// AdaConvBlock_16527034155139
// MI455X (gfx1250) — matrix operand order repaired and checked by reading, not compiled, not run on hardware
//
#include <hip/hip_runtime.h>

// ---------------------------------------------------------------------------
// MI455X / gfx1250 implementation of the AdaConv block.
// All GEMM-like work (kernel predictor + all convolutions) runs on the
// wave32 WMMA pipe: v_wmma_f32_16x16x32_bf16 (bf16 in, f32 accumulate).
// Conv kernel uses a 32-pixel x 64-cout macro-tile per wave (2 A-frags x
// 4 B-frags -> 8 WMMAs per tap/cin-chunk) for 4x activation reuse.
// ---------------------------------------------------------------------------

typedef __attribute__((ext_vector_type(16))) __bf16 v16bf;
typedef __attribute__((ext_vector_type(8)))  __bf16 v8bf;
typedef __attribute__((ext_vector_type(8)))  float  v8f;

#define SHUF16(a,b) __builtin_shufflevector((a),(b),0,1,2,3,4,5,6,7,8,9,10,11,12,13,14,15)

#define B_   8
#define H_   64
#define W_   64
#define CIN_ 256
#define COUT_ 128
#define NG_  4
#define SD_  512
#define HW_  (H_*W_)

__device__ __forceinline__ int refl1(int i, int n) {           // reflect pad=1
    return i < 0 ? -i : (i >= n ? 2*n - 2 - i : i);
}

// ------------------------- small utility kernels ---------------------------

__global__ void k_cvt_bf16(const float* __restrict__ s, __bf16* __restrict__ d, int n) {
    int i = blockIdx.x * blockDim.x + threadIdx.x;
    if (i < n) d[i] = (__bf16)s[i];
}

// w_pool[b,c] = mean of W[b, :2, :2, c]   (W: [B,3,3,SD])
__global__ void k_wpool(const float* __restrict__ W, float* __restrict__ wp) {
    int i = blockIdx.x * blockDim.x + threadIdx.x;    // B*SD
    if (i >= B_*SD_) return;
    int b = i / SD_, c = i % SD_;
    const float* base = W + ((long)b*9)*SD_ + c;
    float v = base[0] + base[SD_] + base[3*SD_] + base[4*SD_];
    wp[i] = v * 0.25f;
}

// pwk[b,n] = wpool[b,:] @ pw_pred_w[:,n] + pw_pred_b[n]   (N = 16384)
__global__ void k_pwk(const float* __restrict__ wp, const float* __restrict__ w,
                      const float* __restrict__ bias, float* __restrict__ out) {
    int i = blockIdx.x * blockDim.x + threadIdx.x;    // B*16384
    if (i >= B_*16384) return;
    int b = i >> 14, n = i & 16383;
    const float* wpb = wp + b*SD_;
    float acc = bias[n];
    for (int k = 0; k < SD_; ++k) acc += wpb[k] * w[(long)k*16384 + n];
    out[i] = acc;
}

// biases[b,c] = wpool[b,:] @ bias_pred_w[:,c] + bias_pred_b[c]
__global__ void k_biaspred(const float* __restrict__ wp, const float* __restrict__ w,
                           const float* __restrict__ bias, float* __restrict__ out) {
    int i = blockIdx.x * blockDim.x + threadIdx.x;    // B*CIN
    if (i >= B_*CIN_) return;
    int b = i >> 8, c = i & 255;
    const float* wpb = wp + b*SD_;
    float acc = bias[c];
    for (int k = 0; k < SD_; ++k) acc += wpb[k] * w[k*CIN_ + c];
    out[i] = acc;
}

// ------------------- kernel predictor (WMMA GEMM) --------------------------
// dwk[b, m(=9 taps), n(=16384)] = im2col(reflect_pad(W)) @ dw_pred_w + b
// M tile = 16 rows (rows 0..8 valid), K = 9*512, N tile = 16 per wave.
// Memory-bound on the 302MB dw_pred_w stream; B is converted f32->bf16 on the fly.
__global__ __launch_bounds__(32) void k_dwk_wmma(
    const __bf16* __restrict__ Wbf,        // [B,3,3,SD] bf16
    const float*  __restrict__ pw,         // [9*SD, 16384] f32
    const float*  __restrict__ pb,         // [16384]
    float*        __restrict__ dwk)        // [B,9,16384]
{
    int lane = threadIdx.x;
    int nt   = blockIdx.x;                 // 0..1023
    int b    = blockIdx.y;
    int m    = lane & 15;
    int me   = (m < 9) ? m : 0;            // rows 9..15 are discarded
    int oy   = me / 3, ox = me % 3;
    int klo  = (lane >> 4) * 16;
    int col  = nt * 16 + (lane & 15);

    v8f acc = {};
    #pragma unroll
    for (int f = 0; f < 9; ++f) {
        int fy = f / 3, fx = f % 3;
        int y = refl1(oy + fy - 1, 3);     // reflect-padded 5x5 style map
        int x = refl1(ox + fx - 1, 3);
        const __bf16* ap = Wbf + (((long)(b*3 + y)*3 + x) * SD_) + klo;
        #pragma unroll 4
        for (int cc = 0; cc < 16; ++cc) {
            int c0 = cc * 32;
            v8bf lo = *(const v8bf*)(ap + c0);
            v8bf hi = *(const v8bf*)(ap + c0 + 8);
            v16bf a = SHUF16(lo, hi);
            const float* wp = pw + ((long)(f*SD_ + c0 + ((lane >> 4) * 16))) * 16384 + col;
            v16bf bb;
            #pragma unroll
            for (int j = 0; j < 16; ++j) bb[j] = (__bf16)wp[(long)j * 16384];
            acc = __builtin_amdgcn_wmma_f32_16x16x32_bf16(
                      false, a, false, bb, (short)0, acc, false, false);
        }
    }
    float bv = pb[col];
    #pragma unroll
    for (int r = 0; r < 8; ++r) {
        int M = r + ((lane >> 4) * 8);
        if (M < 9) dwk[((long)(b*9 + M)) * 16384 + col] = acc[r] + bv;
    }
}

// ------------------- weight packing into B-fragment order ------------------
// src: f32 [K9][CI][CO]  ->  dst: bf16 [nt][k9][cc][lane][16]
//   ci = cc*32 + (lane>>4)*8 + (j&7) + (j>>3)*16 ,  co = nt*16 + (lane&15)
__global__ void k_pack(const float* __restrict__ src, __bf16* __restrict__ dst,
                       int K9, int CI, int CO, long srcBS, long dstBS) {
    int b = blockIdx.y;
    src += b * srcBS; dst += b * dstBS;
    int total = K9 * CI * CO;
    int ccN = CI >> 5;
    for (int idx = blockIdx.x * blockDim.x + threadIdx.x; idx < total;
         idx += gridDim.x * blockDim.x) {
        int j    = idx & 15;
        int lane = (idx >> 4) & 31;
        int t    = idx >> 9;
        int cc   = t % ccN; t /= ccN;
        int k9   = t % K9;
        int nt   = t / K9;
        int ci = cc * 32 + ((lane >> 4) * 8) + (j & 7) + ((j >> 3) * 16);
        int co = nt * 16 + (lane & 15);
        dst[idx] = (__bf16)src[((long)(k9 * CI + ci)) * CO + co];
    }
}

// ------------------------- instance norm -----------------------------------

__global__ __launch_bounds__(256) void k_in_stats(const float* __restrict__ x,
                                                  float* __restrict__ mean,
                                                  float* __restrict__ rstd) {
    __shared__ float ss[256], sq[256];
    int bc = blockIdx.x;                   // b*CIN + c
    int c = bc & 255, b = bc >> 8;
    int t = threadIdx.x;
    float s = 0.f, q = 0.f;
    for (int i = t; i < HW_; i += 256) {
        float v = x[((long)b * HW_ + i) * CIN_ + c];
        s += v; q += v * v;
    }
    ss[t] = s; sq[t] = q; __syncthreads();
    for (int o = 128; o; o >>= 1) {
        if (t < o) { ss[t] += ss[t+o]; sq[t] += sq[t+o]; }
        __syncthreads();
    }
    if (t == 0) {
        float m = ss[0] * (1.f / HW_);
        float v = sq[0] * (1.f / HW_) - m * m;
        mean[bc] = m;
        rstd[bc] = rsqrtf(v + 0.001f);
    }
}

__global__ void k_normalize(const float* __restrict__ x, const float* __restrict__ mean,
                            const float* __restrict__ rstd, __bf16* __restrict__ xn) {
    long i = (long)blockIdx.x * blockDim.x + threadIdx.x;   // B*HW*CIN
    int c = (int)(i & 255);
    int b = (int)(i >> 20);                                 // / (HW*CIN)
    int bc = b * CIN_ + c;
    xn[i] = (__bf16)((x[i] - mean[bc]) * rstd[bc]);
}

// ------------------------- WMMA convolution --------------------------------
// One wave = 32 pixels (two 16-pixel row segments) x NT*16 output channels.
// Per (tap, cin-chunk): 2 A-fragment loads + NT B-fragment loads feed 2*NT
// WMMAs (A reused NT times, B reused 2 times).
// K9 = taps (9 or 1), CC = cin chunks of 32.  Reflect padding, H=W=64.
// Grouped convs: blockIdx.z == group when NT==4, cinBase = z*groupCinStep.
template<int K9, int CC, int NT, int OUTF32, int LEAKY>
__global__ __launch_bounds__(32) void k_conv_wmma(
    const __bf16* __restrict__ in,       // [B,H,W,cinStride] bf16
    const __bf16* __restrict__ wfrag,    // packed fragments
    const float*  __restrict__ bias,     // may be null
    void*         __restrict__ outv,
    int cinStride, int coutStride, int groupCinStep,
    long wBatchStride, int biasBatchStride, float slope)
{
    int lane  = threadIdx.x;
    int wbase = (blockIdx.x & 1) * 32;    // two 32-pixel segments per row
    int h     = blockIdx.x >> 1;
    int b     = blockIdx.y;
    int nt0   = blockIdx.z * NT;
    int cinBase = blockIdx.z * groupCinStep;

    const __bf16* wp = wfrag + (long)b * wBatchStride;
    int m   = lane & 15;                  // pixel within 16-tile (= A row)
    int klo = (lane >> 4) * 8;

    v8f acc[2][NT];
    #pragma unroll
    for (int i = 0; i < 2; ++i)
        #pragma unroll
        for (int n = 0; n < NT; ++n) acc[i][n] = (v8f){};

    for (int k = 0; k < K9; ++k) {
        int dy = (K9 == 9) ? (k / 3) - 1 : 0;
        int dx = (K9 == 9) ? (k % 3) - 1 : 0;
        int hs  = refl1(h + dy, H_);
        int ws0 = refl1(wbase + m + dx, W_);
        int ws1 = refl1(wbase + 16 + m + dx, W_);
        const __bf16* row = in + ((long)(b * H_ + hs)) * W_ * cinStride
                               + cinBase + klo;
        const __bf16* ap0 = row + (long)ws0 * cinStride;
        const __bf16* ap1 = row + (long)ws1 * cinStride;
        #pragma unroll
        for (int cc = 0; cc < CC; ++cc) {
            v16bf a0 = SHUF16(*(const v8bf*)(ap0 + cc*32),
                              *(const v8bf*)(ap0 + cc*32 + 16));
            v16bf a1 = SHUF16(*(const v8bf*)(ap1 + cc*32),
                              *(const v8bf*)(ap1 + cc*32 + 16));
            #pragma unroll
            for (int n = 0; n < NT; ++n) {
                const __bf16* bp = wp + ((long)(nt0 + n) * K9 * CC
                                         + k * CC + cc) * 512 + lane * 16;
                v16bf bb = SHUF16(*(const v8bf*)(bp), *(const v8bf*)(bp + 8));
                acc[0][n] = __builtin_amdgcn_wmma_f32_16x16x32_bf16(
                                false, a0, false, bb, (short)0, acc[0][n], false, false);
                acc[1][n] = __builtin_amdgcn_wmma_f32_16x16x32_bf16(
                                false, a1, false, bb, (short)0, acc[1][n], false, false);
            }
        }
    }

    #pragma unroll
    for (int n = 0; n < NT; ++n) {
        int co = (nt0 + n) * 16 + (lane & 15);
        float bv = (bias != nullptr) ? bias[(long)b * biasBatchStride + co] : 0.f;
        #pragma unroll
        for (int mt = 0; mt < 2; ++mt) {
            #pragma unroll
            for (int r = 0; r < 8; ++r) {
                int pix = wbase + mt * 16 + r + ((lane >> 4) * 8);
                float v = acc[mt][n][r] + bv;
                if (LEAKY) v = v > 0.f ? v : v * slope;
                long oidx = (((long)(b * H_ + h)) * W_ + pix) * coutStride + co;
                if (OUTF32) ((float*)outv)[oidx] = v;
                else        ((__bf16*)outv)[oidx] = (__bf16)v;
            }
        }
    }
}

// ------------------------- bilinear 2x upsample ----------------------------
// y: [B,64,64,COUT] f32 -> out: [B,128,128,COUT] f32 (half-pixel centers)
__global__ void k_upsample(const float* __restrict__ y, float* __restrict__ out) {
    long i = (long)blockIdx.x * blockDim.x + threadIdx.x;  // B*128*128*COUT
    int c  = (int)(i & 127);
    long t = i >> 7;
    int ox = (int)(t & 127); t >>= 7;
    int oy = (int)(t & 127); t >>= 7;
    int b  = (int)t;
    float fy = oy * 0.5f - 0.25f;
    float fx = ox * 0.5f - 0.25f;
    int y0 = (int)floorf(fy), x0 = (int)floorf(fx);
    float wy = fy - (float)y0, wx = fx - (float)x0;
    int y1 = min(y0 + 1, H_ - 1), x1 = min(x0 + 1, W_ - 1);
    y0 = max(y0, 0); x0 = max(x0, 0);
    const float* base = y + (long)b * HW_ * COUT_;
    float v00 = base[((long)y0 * W_ + x0) * COUT_ + c];
    float v01 = base[((long)y0 * W_ + x1) * COUT_ + c];
    float v10 = base[((long)y1 * W_ + x0) * COUT_ + c];
    float v11 = base[((long)y1 * W_ + x1) * COUT_ + c];
    float v0 = v00 + (v01 - v00) * wx;
    float v1 = v10 + (v11 - v10) * wx;
    out[i] = v0 + (v1 - v0) * wy;
}

// ---------------------------------------------------------------------------

extern "C" void kernel_launch(void* const* d_in, const int* in_sizes, int n_in,
                              void* d_out, int out_size, void* d_ws, size_t ws_size,
                              hipStream_t stream) {
    (void)in_sizes; (void)n_in; (void)out_size; (void)ws_size;

    const float* x      = (const float*)d_in[0];   // [8,64,64,256]
    const float* Wsty   = (const float*)d_in[1];   // [8,3,3,512]
    const float* dwpw   = (const float*)d_in[2];   // [3,3,512,16384]
    const float* dwpb   = (const float*)d_in[3];   // [16384]
    const float* pwpw   = (const float*)d_in[4];   // [512,16384]
    const float* pwpb   = (const float*)d_in[5];   // [16384]
    const float* bpw    = (const float*)d_in[6];   // [512,256]
    const float* bpb    = (const float*)d_in[7];   // [256]
    const float* adaw   = (const float*)d_in[8];   // [3,3,256,256]
    const float* adab   = (const float*)d_in[9];   // [256]
    const float* c0w    = (const float*)d_in[10];  // [3,3,256,256]
    const float* c0b    = (const float*)d_in[11];  // [256]
    const float* c1w    = (const float*)d_in[12];  // [3,3,256,128]
    const float* c1b    = (const float*)d_in[13];  // [128]

    // ---- workspace bump allocator (256B aligned) ----
    char* p = (char*)d_ws;
    auto alloc = [&](size_t bytes) -> char* {
        char* r = p; p += (bytes + 255) & ~(size_t)255; return r;
    };
    __bf16* Wbf      = (__bf16*)alloc((size_t)B_*9*SD_ * 2);          // 73.7 KB
    float*  wpool    = (float* )alloc((size_t)B_*SD_ * 4);
    float*  dwkf     = (float* )alloc((size_t)B_*9*16384 * 4);        // 4.7 MB
    float*  pwkf     = (float* )alloc((size_t)B_*16384 * 4);
    float*  biases   = (float* )alloc((size_t)B_*CIN_ * 4);
    float*  meanb    = (float* )alloc((size_t)B_*CIN_ * 4);
    float*  rstdb    = (float* )alloc((size_t)B_*CIN_ * 4);
    __bf16* dwk_pk   = (__bf16*)alloc((size_t)B_*9*64*256 * 2);       // 2.36 MB
    __bf16* pwk_pk   = (__bf16*)alloc((size_t)B_*64*256 * 2);
    __bf16* ada_pk   = (__bf16*)alloc((size_t)9*256*256 * 2);         // 1.18 MB
    __bf16* c0_pk    = (__bf16*)alloc((size_t)9*256*256 * 2);
    __bf16* c1_pk    = (__bf16*)alloc((size_t)9*256*128 * 2);
    __bf16* actA     = (__bf16*)alloc((size_t)B_*HW_*CIN_ * 2);       // 16.8 MB
    __bf16* actB     = (__bf16*)alloc((size_t)B_*HW_*CIN_ * 2);
    __bf16* actC     = (__bf16*)alloc((size_t)B_*HW_*CIN_ * 2);
    float*  y4       = (float* )alloc((size_t)B_*HW_*COUT_ * 4);      // 16.8 MB

    // ---- kernel predictor ----
    k_cvt_bf16<<<(B_*9*SD_ + 255)/256, 256, 0, stream>>>(Wsty, Wbf, B_*9*SD_);
    k_wpool   <<<(B_*SD_ + 255)/256, 256, 0, stream>>>(Wsty, wpool);
    k_pwk     <<<(B_*16384)/256, 256, 0, stream>>>(wpool, pwpw, pwpb, pwkf);
    k_biaspred<<<(B_*CIN_)/256, 256, 0, stream>>>(wpool, bpw, bpb, biases);
    k_dwk_wmma<<<dim3(1024, B_), 32, 0, stream>>>(Wbf, dwpw, dwpb, dwkf);

    // ---- pack all weights into WMMA B-fragment order ----
    k_pack<<<dim3(576, B_), 256, 0, stream>>>(dwkf, dwk_pk, 9, 64, 256,
                                              (long)9*64*256, (long)9*64*256);
    k_pack<<<dim3(64, B_), 256, 0, stream>>>(pwkf, pwk_pk, 1, 64, 256,
                                             (long)64*256, (long)64*256);
    k_pack<<<dim3(2304, 1), 256, 0, stream>>>(adaw, ada_pk, 9, 256, 256, 0, 0);
    k_pack<<<dim3(2304, 1), 256, 0, stream>>>(c0w, c0_pk, 9, 256, 256, 0, 0);
    k_pack<<<dim3(1152, 1), 256, 0, stream>>>(c1w, c1_pk, 9, 256, 128, 0, 0);

    // ---- instance norm -> bf16 activations ----
    k_in_stats<<<B_*CIN_, 256, 0, stream>>>(x, meanb, rstdb);
    k_normalize<<<(B_*HW_*CIN_)/256, 256, 0, stream>>>(x, meanb, rstdb, actA);

    dim3 cgrid(H_ * (W_/32), B_, 4);    // 4 z-blocks of 4 cout tiles (256 outs)
    // grouped 3x3 (per-sample dwk):  actA -> actB   (z == group)
    k_conv_wmma<9, 2, 4, 0, 0><<<cgrid, 32, 0, stream>>>(
        actA, dwk_pk, nullptr, actB, CIN_, CIN_, 64,
        (long)16*9*2*512, 0, 0.f);
    // grouped 1x1 (per-sample pwk) + per-sample bias: actB -> actC
    k_conv_wmma<1, 2, 4, 0, 0><<<cgrid, 32, 0, stream>>>(
        actB, pwk_pk, biases, actC, CIN_, CIN_, 64,
        (long)16*1*2*512, CIN_, 0.f);
    // static 3x3 ada_conv + bias: actC -> actA
    k_conv_wmma<9, 8, 4, 0, 0><<<cgrid, 32, 0, stream>>>(
        actC, ada_pk, adab, actA, CIN_, CIN_, 0,
        0, 0, 0.f);
    // conv0 + bias + leaky(0.2): actA -> actB
    k_conv_wmma<9, 8, 4, 0, 1><<<cgrid, 32, 0, stream>>>(
        actA, c0_pk, c0b, actB, CIN_, CIN_, 0,
        0, 0, 0.2f);
    // conv1 + bias + leaky(0.2), f32 out: actB -> y4
    dim3 cgrid1(H_ * (W_/32), B_, 2);   // 2 z-blocks of 4 tiles (128 outs)
    k_conv_wmma<9, 8, 4, 1, 1><<<cgrid1, 32, 0, stream>>>(
        actB, c1_pk, c1b, y4, CIN_, COUT_, 0,
        0, 0, 0.2f);

    // ---- bilinear x2 upsample -> d_out ----
    k_upsample<<<(B_*128*128*COUT_)/256, 256, 0, stream>>>(y4, (float*)d_out);
}
